// Attention_16174846837080
// MI455X (gfx1250) — compile-verified, matrix operand order checked by reading, not run on hardware
//
#include <hip/hip_runtime.h>
#include <hip/hip_bf16.h>

// ---------- types ----------
typedef __attribute__((ext_vector_type(16))) __bf16 v16bf;
typedef __attribute__((ext_vector_type(8)))  float  v8f;
typedef int v4i __attribute__((vector_size(16)));   // matches builtin's int4 pointee

union FragB {
    v16bf v;
    unsigned u[8];
    unsigned short s[16];
};

// ---------- async global->LDS staging (CDNA5 GLOBAL_LOAD_ASYNC_TO_LDS_B128) ----------
#if defined(__has_builtin)
#if __has_builtin(__builtin_amdgcn_global_load_async_to_lds_b128) && \
    __has_builtin(__builtin_amdgcn_s_wait_asynccnt)
#define USE_ASYNC_LDS 1
#endif
#endif
#ifndef USE_ASYNC_LDS
#define USE_ASYNC_LDS 0
#endif

typedef __attribute__((address_space(1))) v4i gv4i;
typedef __attribute__((address_space(3))) v4i lv4i;

__device__ __forceinline__ void stage16(const unsigned short* __restrict__ g,
                                        unsigned short* l) {
#if USE_ASYNC_LDS
    __builtin_amdgcn_global_load_async_to_lds_b128((gv4i*)g, (lv4i*)l, 0, 0);
#else
    *(uint4*)l = *(const uint4*)g;
#endif
}

__device__ __forceinline__ void stage_commit() {
#if USE_ASYNC_LDS
    __builtin_amdgcn_s_wait_asynccnt(0);
#endif
    __syncthreads();
}

// ---------- helpers ----------
__device__ __forceinline__ unsigned short f2bf(float f) {
    unsigned u = __float_as_uint(f);
    unsigned r = u + 0x7FFFu + ((u >> 16) & 1u);   // round-to-nearest-even
    return (unsigned short)(r >> 16);
}

// CDNA5 16-bit A-matrix 16x32 layout: lane L holds row m=L&15; half=L>>4.
// VGPR p (0..7) holds K pair starting at:
__device__ __forceinline__ int koff(int p, int half) {
    return ((p >> 2) << 4) + (half << 3) + ((p & 3) << 1);
}

__device__ __forceinline__ float hmax16(float v) {
    v = fmaxf(v, __shfl_xor(v, 1));
    v = fmaxf(v, __shfl_xor(v, 2));
    v = fmaxf(v, __shfl_xor(v, 4));
    v = fmaxf(v, __shfl_xor(v, 8));
    return v;
}
__device__ __forceinline__ float hsum16(float v) {
    v += __shfl_xor(v, 1);
    v += __shfl_xor(v, 2);
    v += __shfl_xor(v, 4);
    v += __shfl_xor(v, 8);
    return v;
}

// ---------- constants ----------
#define BATCH 64
#define NSEQ  577
#define CDIM  768
#define HEADS 12
#define DHEAD 64
#define KTILE 32

// ---------- f32 -> bf16 conversion ----------
__global__ __launch_bounds__(256) void cvt_kernel(const float* __restrict__ src,
                                                  unsigned short* __restrict__ dst,
                                                  long n) {
    long i = (long)blockIdx.x * blockDim.x + threadIdx.x;
    long stride = (long)gridDim.x * blockDim.x;
    for (; i < n; i += stride) dst[i] = f2bf(src[i]);
}

// ---------- shared GEMM mainloop: Y(64x64) = A(64xK) @ W(64xK)^T ----------
__device__ __forceinline__ void gemm_tile(const unsigned short* __restrict__ A, int aRows, int lda,
                                          const unsigned short* __restrict__ W, int ldw, int oBase,
                                          int mBase, int K,
                                          unsigned short* As, unsigned short* Ws_,
                                          v8f acc[4]) {
    const int tid  = threadIdx.x;
    const int lane = tid & 31;
    const int wave = tid >> 5;
    const int half = lane >> 4;
    const int lh   = lane & 15;

    for (int k0 = 0; k0 < K; k0 += KTILE) {
        // cooperative stage: A tile 64x32 bf16 and W tile 64x32 bf16 (16B chunks, async->LDS)
        for (int c = tid; c < 256; c += 128) {
            int r  = c >> 2;
            int k8 = (c & 3) << 3;
            int gm = mBase + r;
            if (gm > aRows - 1) gm = aRows - 1;   // clamp: garbage rows never stored
            stage16(A + (size_t)gm * lda + k0 + k8, As + r * KTILE + k8);
            int go = oBase + r;
            stage16(W + (size_t)go * ldw + k0 + k8, Ws_ + r * KTILE + k8);
        }
        // prefetch next K tile
        int kn = k0 + KTILE;
        if (kn < K) {
            int r  = tid >> 1;
            int gm = mBase + r;
            if (gm > aRows - 1) gm = aRows - 1;
            __builtin_prefetch(A + (size_t)gm * lda + kn + (tid & 1) * 16, 0, 1);
            __builtin_prefetch(W + (size_t)(oBase + r) * ldw + kn + (tid & 1) * 16, 0, 1);
        }
        stage_commit();

        FragB a;
#pragma unroll
        for (int p = 0; p < 8; ++p)
            a.u[p] = *(const unsigned*)(As + (wave * 16 + lh) * KTILE + koff(p, half));

#pragma unroll
        for (int f = 0; f < 4; ++f) {
            FragB bf;
#pragma unroll
            for (int p = 0; p < 8; ++p)
                bf.u[p] = *(const unsigned*)(Ws_ + (f * 16 + lh) * KTILE + koff(p, half));
            acc[f] = __builtin_amdgcn_wmma_f32_16x16x32_bf16(
                false, a.v, false, bf.v, (short)0, acc[f], false, false);
        }
        __syncthreads();
    }
}

// ---------- QKV GEMM: qkv[3,B,H,N,D] (bf16) = x @ qkv_w^T, small-branch head zeroing ----------
__global__ __launch_bounds__(128) void gemm_qkv_kernel(const unsigned short* __restrict__ xbf,
                                                       const unsigned short* __restrict__ wq,
                                                       unsigned short* __restrict__ qkvb) {
    __shared__ __align__(16) unsigned short As[64 * KTILE];
    __shared__ __align__(16) unsigned short Ws_[64 * KTILE];
    const int b = blockIdx.z, ot = blockIdx.y, mt = blockIdx.x;

    v8f acc[4] = {};
    gemm_tile(xbf + (size_t)b * NSEQ * CDIM, NSEQ, CDIM,
              wq, CDIM, ot * 64, mt * 64, CDIM, As, Ws_, acc);

    const int lane = threadIdx.x & 31, wave = threadIdx.x >> 5;
    const int half = lane >> 4, lh = lane & 15;
#pragma unroll
    for (int f = 0; f < 4; ++f) {
        int go    = ot * 64 + f * 16 + lh;      // [0, 2304)
        int which = go / CDIM;                  // 0=q 1=k 2=v
        int cc    = go % CDIM;
        int head  = cc >> 6;
        int dd    = cc & 63;
        bool zero = (b >= BATCH / 2) && (head >= HEADS / 2);
#pragma unroll
        for (int v = 0; v < 8; ++v) {
            int gm = mt * 64 + wave * 16 + v + 8 * half;
            if (gm < NSEQ) {
                float val = zero ? 0.f : acc[f][v];
                qkvb[((((size_t)which * BATCH + b) * HEADS + head) * NSEQ + gm) * DHEAD + dd] =
                    f2bf(val);
            }
        }
    }
}

// ---------- Flash attention: per (row-tile, head, batch) ----------
__global__ __launch_bounds__(128) void attn_kernel(const unsigned short* __restrict__ qkvb,
                                                   unsigned short* __restrict__ aout) {
    __shared__ __align__(16) unsigned short ks[32 * 64];      // keys x d
    __shared__ __align__(16) unsigned short vs[32 * 64];      // keys x d (row-major)
    __shared__ __align__(16) unsigned short pls[4][16 * 32];  // per-wave P staging

    const int b = blockIdx.z, h = blockIdx.y, mt = blockIdx.x;
    const int tid = threadIdx.x, lane = tid & 31, wave = tid >> 5;
    const int half = lane >> 4, lh = lane & 15;
    const size_t hs = (size_t)NSEQ * DHEAD;

    const unsigned short* qp = qkvb + (((size_t)0 * BATCH + b) * HEADS + h) * hs;
    const unsigned short* kp = qkvb + (((size_t)1 * BATCH + b) * HEADS + h) * hs;
    const unsigned short* vp = qkvb + (((size_t)2 * BATCH + b) * HEADS + h) * hs;

    // q fragments (2 K-chunks of 32 covering D=64), held in registers for all key blocks
    FragB qf[2];
    {
        int gm = mt * 64 + wave * 16 + lh;
        if (gm > NSEQ - 1) gm = NSEQ - 1;
#pragma unroll
        for (int ch = 0; ch < 2; ++ch)
#pragma unroll
            for (int p = 0; p < 8; ++p)
                qf[ch].u[p] = *(const unsigned*)(qp + (size_t)gm * DHEAD + ch * 32 + koff(p, half));
    }

    v8f Oacc[4] = {};
    float mrun[8], lrun[8];
#pragma unroll
    for (int v = 0; v < 8; ++v) { mrun[v] = -3.0e38f; lrun[v] = 0.f; }

    for (int j0 = 0; j0 < NSEQ; j0 += 32) {
        // stage K and V tiles (each 32 keys x 64 d, bf16) via async->LDS
        for (int c = tid; c < 512; c += 128) {
            int t   = c >> 8;                 // 0 = K, 1 = V
            int cc  = c & 255;
            int key = cc >> 3, d8 = (cc & 7) << 3;
            int gk = j0 + key;
            if (gk > NSEQ - 1) gk = NSEQ - 1; // clamp: masked / P==0 contributions
            const unsigned short* src = (t ? vp : kp) + (size_t)gk * DHEAD + d8;
            stage16(src, (t ? vs : ks) + key * DHEAD + d8);
        }
        // prefetch next key block
        int jn = j0 + 32;
        if (jn < NSEQ) {
            int pk = jn + (tid >> 2);
            if (pk > NSEQ - 1) pk = NSEQ - 1;
            __builtin_prefetch(kp + (size_t)pk * DHEAD + (tid & 3) * 16, 0, 1);
            __builtin_prefetch(vp + (size_t)pk * DHEAD + (tid & 3) * 16, 0, 1);
        }
        stage_commit();

        // S = q @ k^T for two 16-key sub-tiles
        float S[2][8];
#pragma unroll
        for (int jj = 0; jj < 2; ++jj) {
            FragB kb0, kb1;
#pragma unroll
            for (int p = 0; p < 8; ++p) {
                kb0.u[p] = *(const unsigned*)(ks + (jj * 16 + lh) * DHEAD + koff(p, half));
                kb1.u[p] = *(const unsigned*)(ks + (jj * 16 + lh) * DHEAD + 32 + koff(p, half));
            }
            v8f s = {};
            s = __builtin_amdgcn_wmma_f32_16x16x32_bf16(false, qf[0].v, false, kb0.v, (short)0, s, false, false);
            s = __builtin_amdgcn_wmma_f32_16x16x32_bf16(false, qf[1].v, false, kb1.v, (short)0, s, false, false);
            int key = j0 + jj * 16 + lh;
#pragma unroll
            for (int v = 0; v < 8; ++v) {
                float sv = s[v] * 0.125f;           // D^-0.5
                if (key >= NSEQ) sv = -3.0e38f;     // mask OOB keys (overwrites any value)
                S[jj][v] = sv;
            }
        }

        // online softmax (row = VGPR index + 8*half; 16-lane butterfly per half)
        float corr[8];
#pragma unroll
        for (int v = 0; v < 8; ++v) {
            float tmax = hmax16(fmaxf(S[0][v], S[1][v]));
            float mnew = fmaxf(mrun[v], tmax);
            corr[v] = __expf(mrun[v] - mnew);
            float p0 = __expf(S[0][v] - mnew);
            float p1 = __expf(S[1][v] - mnew);
            lrun[v] = lrun[v] * corr[v] + hsum16(p0 + p1);
            mrun[v] = mnew;
            int r = v + 8 * half;
            pls[wave][r * 32 + lh]      = f2bf(p0);
            pls[wave][r * 32 + 16 + lh] = f2bf(p1);
        }
        __syncthreads();

        // P as A-fragment, V as B-fragments; O = corr*O + P @ V
        FragB pa;
#pragma unroll
        for (int p = 0; p < 8; ++p)
            pa.u[p] = *(const unsigned*)(&pls[wave][lh * 32 + koff(p, half)]);
#pragma unroll
        for (int f = 0; f < 4; ++f) {
            FragB vb;
            int d = f * 16 + lh;
#pragma unroll
            for (int p = 0; p < 8; ++p) {
                int kk = koff(p, half);
                vb.s[2 * p]     = vs[(size_t)kk * DHEAD + d];
                vb.s[2 * p + 1] = vs[(size_t)(kk + 1) * DHEAD + d];
            }
            v8f o = Oacc[f];
#pragma unroll
            for (int v = 0; v < 8; ++v) o[v] *= corr[v];
            Oacc[f] = __builtin_amdgcn_wmma_f32_16x16x32_bf16(
                false, pa.v, false, vb.v, (short)0, o, false, false);
        }
        __syncthreads();
    }

    // normalize and write [B,N,C] bf16 (c = h*64 + d)
#pragma unroll
    for (int f = 0; f < 4; ++f) {
#pragma unroll
        for (int v = 0; v < 8; ++v) {
            int gm = mt * 64 + wave * 16 + v + 8 * half;
            if (gm < NSEQ) {
                float val = Oacc[f][v] / lrun[v];
                aout[((size_t)b * NSEQ + gm) * CDIM + h * DHEAD + f * 16 + lh] = f2bf(val);
            }
        }
    }
}

// ---------- Proj GEMM: out = attn_out @ proj_w^T + b, small-branch channel zeroing ----------
__global__ __launch_bounds__(128) void gemm_proj_kernel(const unsigned short* __restrict__ abf,
                                                        const unsigned short* __restrict__ wp,
                                                        const float* __restrict__ bias,
                                                        float* __restrict__ out) {
    __shared__ __align__(16) unsigned short As[64 * KTILE];
    __shared__ __align__(16) unsigned short Ws_[64 * KTILE];
    const int b = blockIdx.z, ot = blockIdx.y, mt = blockIdx.x;

    v8f acc[4] = {};
    gemm_tile(abf + (size_t)b * NSEQ * CDIM, NSEQ, CDIM,
              wp, CDIM, ot * 64, mt * 64, CDIM, As, Ws_, acc);

    const int lane = threadIdx.x & 31, wave = threadIdx.x >> 5;
    const int half = lane >> 4, lh = lane & 15;
#pragma unroll
    for (int f = 0; f < 4; ++f) {
        int go = ot * 64 + f * 16 + lh;
        float bv = bias[go];
        bool zero = (b >= BATCH / 2) && (go >= CDIM / 2);
#pragma unroll
        for (int v = 0; v < 8; ++v) {
            int gm = mt * 64 + wave * 16 + v + 8 * half;
            if (gm < NSEQ) {
                float val = zero ? 0.f : (acc[f][v] + bv);
                out[((size_t)b * NSEQ + gm) * CDIM + go] = val;
            }
        }
    }
}

// ---------- host launch ----------
extern "C" void kernel_launch(void* const* d_in, const int* in_sizes, int n_in,
                              void* d_out, int out_size, void* d_ws, size_t ws_size,
                              hipStream_t stream) {
    (void)in_sizes; (void)n_in; (void)out_size; (void)ws_size;
    const float* x      = (const float*)d_in[0];
    const float* qkv_w  = (const float*)d_in[1];
    const float* proj_w = (const float*)d_in[2];
    const float* proj_b = (const float*)d_in[3];
    float* out = (float*)d_out;

    unsigned short* ws = (unsigned short*)d_ws;
    const size_t S_x = (size_t)BATCH * NSEQ * CDIM;        // 28,360,704
    unsigned short* xbf    = ws;                            // reused as attn_out after QKV GEMM
    unsigned short* qkvwb  = xbf + S_x;
    unsigned short* projwb = qkvwb + (size_t)(3 * CDIM) * CDIM;
    unsigned short* qkvb   = projwb + (size_t)CDIM * CDIM;  // [3,B,H,N,D] bf16

    cvt_kernel<<<2048, 256, 0, stream>>>(x, xbf, (long)S_x);
    cvt_kernel<<<1024, 256, 0, stream>>>(qkv_w, qkvwb, (long)(3 * CDIM) * CDIM);
    cvt_kernel<<<512, 256, 0, stream>>>(proj_w, projwb, (long)CDIM * CDIM);

    dim3 gq(10, (3 * CDIM) / 64, BATCH);   // (10, 36, 64)
    gemm_qkv_kernel<<<gq, 128, 0, stream>>>(xbf, qkvwb, qkvb);

    dim3 ga(10, HEADS, BATCH);             // (10, 12, 64)
    attn_kernel<<<ga, 128, 0, stream>>>(qkvb, xbf);

    dim3 gp(10, CDIM / 64, BATCH);         // (10, 12, 64)
    gemm_proj_kernel<<<gp, 128, 0, stream>>>(xbf, projwb, proj_b, out);
}
